// RTDFRCNNBackBone_34059090657833
// MI455X (gfx1250) — matrix operand order repaired and checked by reading, not compiled, not run on hardware
//
#include <hip/hip_runtime.h>
#include <hip/hip_bf16.h>

typedef _Float16 half_t;
typedef __attribute__((ext_vector_type(16))) _Float16 v16h;
typedef __attribute__((ext_vector_type(8)))  float    v8f;

#define B_    4
#define NV_   4096
#define C_    64
#define H_    376
#define W_    1240

// ---------------------------------------------------------------------------
// WMMA helpers (layouts per CDNA5 ISA 7.12.2, wave32)
// ---------------------------------------------------------------------------
__device__ __forceinline__ v8f wmma_f16(v16h a, v16h b, v8f c) {
    return __builtin_amdgcn_wmma_f32_16x16x32_f16(false, a, false, b, (short)0, c,
                                                  false, false);
}

// A fragment 16x32 f16, row-major source with leading dim ldh (halves).
// lane L: M = L&15 ; K = koff + i (contiguous 16 halves), koff = 16*(L>=16)
__device__ __forceinline__ v16h load_A16x32(const half_t* base, int ldh) {
    int lane = threadIdx.x & 31;
    int m    = lane & 15;
    int koff = (lane >> 4) << 4;
    const half_t* p = base + m * ldh + koff;
    v16h a;
#pragma unroll
    for (int i = 0; i < 8; ++i) a[i] = p[i];
    p += 8;
#pragma unroll
    for (int i = 0; i < 8; ++i) a[8 + i] = p[i];
    return a;
}

// B fragment 32x16 f16 where B[k][n] = base[n*ldh + k]  (i.e. "rows are N").
// lane L: N = L&15 ; K = 16*(L>=16) + i  (contiguous 16 halves)
__device__ __forceinline__ v16h load_B_rows(const half_t* base, int ldh) {
    int lane = threadIdx.x & 31;
    const half_t* p = base + (lane & 15) * ldh + ((lane >> 4) << 4);
    v16h b;
#pragma unroll
    for (int i = 0; i < 16; ++i) b[i] = p[i];
    return b;
}

// Same but source is f32 (weights), converted on load.
__device__ __forceinline__ v16h load_B_rows_f32(const float* base, int ld) {
    int lane = threadIdx.x & 31;
    const float* p = base + (lane & 15) * ld + ((lane >> 4) << 4);
    v16h b;
#pragma unroll
    for (int i = 0; i < 16; ++i) b[i] = (half_t)p[i];
    return b;
}

// ---------------------------------------------------------------------------
// gfx1250 async DMA: global -> LDS, 16B per lane, tracked by ASYNCcnt
// ---------------------------------------------------------------------------
__device__ __forceinline__ void async_b128(unsigned lds_byte, const half_t* gaddr) {
    asm volatile("global_load_async_to_lds_b128 %0, %1, off"
                 :: "v"(lds_byte), "v"(gaddr)
                 : "memory");
}
__device__ __forceinline__ void wait_async0() {
    asm volatile("s_wait_asynccnt 0" ::: "memory");
}
__device__ __forceinline__ unsigned lds_off(const void* p) {
    return (unsigned)(unsigned long long)p;   // generic LDS addr: low 32 bits = offset
}

// ---------------------------------------------------------------------------
// Kernel 1: gather + Q/K/V projections -> fp16 (Q pre-scaled, V transposed)
// block = 256 threads = 4 rows x 64 channels
// ---------------------------------------------------------------------------
__global__ __launch_bounds__(256) void qkv_prologue(
    const float* __restrict__ voxel_features, const float* __restrict__ voxels_3d,
    const float* __restrict__ img_feats, const int* __restrict__ vox_x,
    const int* __restrict__ vox_y,
    const float* __restrict__ q_w, const float* __restrict__ q_b,
    const float* __restrict__ k_w, const float* __restrict__ k_b,
    const float* __restrict__ v_w, const float* __restrict__ v_b,
    half_t* __restrict__ Qh, half_t* __restrict__ Kh, half_t* __restrict__ Vt)
{
    __shared__ float cam[4][C_];
    __shared__ float vf [4][C_];
    int tid  = threadIdx.x;
    int rloc = tid >> 6;
    int c    = tid & 63;
    int r    = blockIdx.x * 4 + rloc;
    int b    = r >> 12;
    int n    = r & (NV_ - 1);

    int x = vox_x[r], y = vox_y[r];
    bool valid = (x >= 0) && (x < W_) && (y >= 0) && (y < H_);
    int xc = min(max(x, 0), W_ - 1);
    int yc = min(max(y, 0), H_ - 1);
    float camv = valid ? img_feats[(((size_t)b * C_ + c) * H_ + yc) * W_ + xc] : 0.0f;
    cam[rloc][c] = camv;
    vf [rloc][c] = voxel_features[(size_t)r * C_ + c];
    __syncthreads();

    float accq = q_b[c];
    float accv = v_b[c];
    const float* qwr = q_w + c * C_;
    const float* vwr = v_w + c * C_;
#pragma unroll 8
    for (int i = 0; i < C_; ++i) {
        accq = fmaf(cam[rloc][i], qwr[i], accq);
        accv = fmaf(vf [rloc][i], vwr[i], accv);
    }
    accq *= 0.125f;   // 1/sqrt(64) folded into Q

    float p0 = voxels_3d[(size_t)r * 3 + 0];
    float p1 = voxels_3d[(size_t)r * 3 + 1];
    float p2 = voxels_3d[(size_t)r * 3 + 2];
    float acck = k_b[c] + k_w[c * 3 + 0] * p0 + k_w[c * 3 + 1] * p1 + k_w[c * 3 + 2] * p2;

    Qh[(size_t)r * C_ + c] = (half_t)accq;
    Kh[(size_t)r * C_ + c] = (half_t)acck;
    Vt[((size_t)b * C_ + c) * NV_ + n] = (half_t)accv;   // V transposed: [B][C][NV]
}

// ---------------------------------------------------------------------------
// Kernel 2: flash attention, 4 waves/block sharing async-DMA'd K/V tiles.
// Block handles 64 consecutive queries of one batch (16 per wave).
// ---------------------------------------------------------------------------
#define WPB 4

__global__ __launch_bounds__(128) void flash_attn_epilogue(
    const half_t* __restrict__ Qh, const half_t* __restrict__ Kh,
    const half_t* __restrict__ Vt,
    const float* __restrict__ voxel_features,
    const float* __restrict__ o_w, const float* __restrict__ o_b,
    const float* __restrict__ fc_w,
    float* __restrict__ out)
{
    __shared__ __align__(16) half_t Ktile[2][32 * 64];   // 4KB per buffer
    __shared__ __align__(16) half_t Vtile[2][64 * 32];   // 4KB per buffer
    __shared__ __align__(16) half_t pstage[WPB][16 * 64];

    int tid  = threadIdx.x;
    int lane = tid & 31;
    int wave = tid >> 5;
    int blk  = blockIdx.x;                       // 0..255
    int b    = blk >> 6;                         // 64 blocks per batch
    int q0   = ((blk & 63) << 6) + (wave << 4);  // this wave's 16-query tile

    half_t* stage = &pstage[wave][0];
    int rbase = (lane >> 4) << 3;                // D-frag: rows j+rbase
    int col   = lane & 15;                       // D-frag: column

    const half_t* Qbase = Qh + ((size_t)b * NV_ + q0) * C_;
    v16h qa0 = load_A16x32(Qbase, C_);           // K dims  0..31
    v16h qa1 = load_A16x32(Qbase + 32, C_);      // K dims 32..63

    v8f o0 = {}, o1 = {}, o2 = {}, o3 = {};
    float mrow[8], lrow[8];
#pragma unroll
    for (int j = 0; j < 8; ++j) { mrow[j] = -1e30f; lrow[j] = 0.0f; }

    const half_t* Kb = Kh + (size_t)b * NV_ * C_;
    const half_t* Vb = Vt + (size_t)b * C_ * NV_;

    // cooperative async prefetch of one 32-key step into buffer `buf`
    auto prefetch = [&](int k0, int buf) {
        // K tile: 32 rows x 64 halves, contiguous 4KB in global
        {
            const half_t* g = Kb + (size_t)k0 * C_;
            unsigned l = lds_off(&Ktile[buf][0]);
#pragma unroll
            for (int i = 0; i < 2; ++i) {
                int off = (tid + 128 * i) * 8;              // halves
                async_b128(l + off * 2, g + off);
            }
        }
        // V tile: 64 channel rows x 32 halves (row stride NV_ in global)
        {
            unsigned l = lds_off(&Vtile[buf][0]);
#pragma unroll
            for (int i = 0; i < 2; ++i) {
                int q = tid + 128 * i;                      // 0..255 16B chunks
                int row = q >> 2, c4 = q & 3;
                async_b128(l + (row * 32 + c4 * 8) * 2,
                           Vb + (size_t)row * NV_ + k0 + c4 * 8);
            }
        }
    };

    prefetch(0, 0);

    for (int step = 0; step < NV_ / 32; ++step) {
        int k0  = step << 5;
        int buf = step & 1;

        wait_async0();        // this wave's share of the DMA for `buf` is done
        __syncthreads();      // all waves' shares done; prev buffer fully read
        if (step + 1 < NV_ / 32) prefetch(k0 + 32, buf ^ 1);

        const half_t* Kt = &Ktile[buf][0];
        const half_t* Vl = &Vtile[buf][0];

        // ---- S = Q K^T : two 16x16 tiles ----
        v8f s0 = {}, s1 = {};
        {
            v16h bk;
            bk = load_B_rows(Kt,               C_); s0 = wmma_f16(qa0, bk, s0);
            bk = load_B_rows(Kt + 32,          C_); s0 = wmma_f16(qa1, bk, s0);
            bk = load_B_rows(Kt + 16 * C_,     C_); s1 = wmma_f16(qa0, bk, s1);
            bk = load_B_rows(Kt + 16 * C_ + 32,C_); s1 = wmma_f16(qa1, bk, s1);
        }
        // ---- online softmax: cross-lane MAX only; row-sum kept per-lane ----
        float alpha[8];
#pragma unroll
        for (int j = 0; j < 8; ++j) {
            float mx = fmaxf(s0[j], s1[j]);
            mx = fmaxf(mx, __shfl_xor(mx, 1));
            mx = fmaxf(mx, __shfl_xor(mx, 2));
            mx = fmaxf(mx, __shfl_xor(mx, 4));
            mx = fmaxf(mx, __shfl_xor(mx, 8));
            float mnew = fmaxf(mrow[j], mx);
            float a  = __expf(mrow[j] - mnew);
            float p0 = __expf(s0[j] - mnew);
            float p1 = __expf(s1[j] - mnew);
            s0[j] = p0; s1[j] = p1;
            lrow[j]  = lrow[j] * a + (p0 + p1);   // per-lane partial sum
            mrow[j]  = mnew;
            alpha[j] = a;
        }
#pragma unroll
        for (int j = 0; j < 8; ++j) {
            o0[j] *= alpha[j]; o1[j] *= alpha[j];
            o2[j] *= alpha[j]; o3[j] *= alpha[j];
        }
        // ---- stage P (16x32) to LDS f16: D-layout -> row-major ----
#pragma unroll
        for (int j = 0; j < 8; ++j) {
            stage[(rbase + j) * 32 + col]      = (half_t)s0[j];
            stage[(rbase + j) * 32 + col + 16] = (half_t)s1[j];
        }
        asm volatile("s_wait_dscnt 0" ::: "memory");
        v16h pa = load_A16x32(stage, 32);
        // ---- O += P @ V(32x64) from the shared V tile ----
        {
            v16h bv;
            bv = load_B_rows(Vl +  0 * 32, 32); o0 = wmma_f16(pa, bv, o0);
            bv = load_B_rows(Vl + 16 * 32, 32); o1 = wmma_f16(pa, bv, o1);
            bv = load_B_rows(Vl + 32 * 32, 32); o2 = wmma_f16(pa, bv, o2);
            bv = load_B_rows(Vl + 48 * 32, 32); o3 = wmma_f16(pa, bv, o3);
        }
    }

    // ---- finalize row sums (one cross-lane reduction) and normalize ----
#pragma unroll
    for (int j = 0; j < 8; ++j) {
        float ls = lrow[j];
        ls += __shfl_xor(ls, 1);
        ls += __shfl_xor(ls, 2);
        ls += __shfl_xor(ls, 4);
        ls += __shfl_xor(ls, 8);
        float r = 1.0f / ls;
        o0[j] *= r; o1[j] *= r; o2[j] *= r; o3[j] *= r;
    }

    // ---- epilogue GEMM 1: T = ctx @ o_w^T + o_b ----
#pragma unroll
    for (int j = 0; j < 8; ++j) {
        stage[(rbase + j) * 64 + col]      = (half_t)o0[j];
        stage[(rbase + j) * 64 + col + 16] = (half_t)o1[j];
        stage[(rbase + j) * 64 + col + 32] = (half_t)o2[j];
        stage[(rbase + j) * 64 + col + 48] = (half_t)o3[j];
    }
    asm volatile("s_wait_dscnt 0" ::: "memory");
    v16h a0 = load_A16x32(stage, 64);
    v16h a1 = load_A16x32(stage + 32, 64);

    v8f t0 = {}, t1 = {}, t2 = {}, t3 = {};
    {
        v16h bw;
        bw = load_B_rows_f32(o_w +  0 * 64,      64); t0 = wmma_f16(a0, bw, t0);
        bw = load_B_rows_f32(o_w +  0 * 64 + 32, 64); t0 = wmma_f16(a1, bw, t0);
        bw = load_B_rows_f32(o_w + 16 * 64,      64); t1 = wmma_f16(a0, bw, t1);
        bw = load_B_rows_f32(o_w + 16 * 64 + 32, 64); t1 = wmma_f16(a1, bw, t1);
        bw = load_B_rows_f32(o_w + 32 * 64,      64); t2 = wmma_f16(a0, bw, t2);
        bw = load_B_rows_f32(o_w + 32 * 64 + 32, 64); t2 = wmma_f16(a1, bw, t2);
        bw = load_B_rows_f32(o_w + 48 * 64,      64); t3 = wmma_f16(a0, bw, t3);
        bw = load_B_rows_f32(o_w + 48 * 64 + 32, 64); t3 = wmma_f16(a1, bw, t3);
    }
    {
        float b0 = o_b[col], b1 = o_b[col + 16], b2 = o_b[col + 32], b3 = o_b[col + 48];
#pragma unroll
        for (int j = 0; j < 8; ++j) { t0[j] += b0; t1[j] += b1; t2[j] += b2; t3[j] += b3; }
    }

    // ---- epilogue GEMM 2: F = relu(T @ fc_w^T) ----
#pragma unroll
    for (int j = 0; j < 8; ++j) {
        stage[(rbase + j) * 64 + col]      = (half_t)t0[j];
        stage[(rbase + j) * 64 + col + 16] = (half_t)t1[j];
        stage[(rbase + j) * 64 + col + 32] = (half_t)t2[j];
        stage[(rbase + j) * 64 + col + 48] = (half_t)t3[j];
    }
    asm volatile("s_wait_dscnt 0" ::: "memory");
    a0 = load_A16x32(stage, 64);
    a1 = load_A16x32(stage + 32, 64);

    v8f f0 = {}, f1 = {}, f2 = {}, f3 = {};
    {
        v16h bw;
        bw = load_B_rows_f32(fc_w +  0 * 64,      64); f0 = wmma_f16(a0, bw, f0);
        bw = load_B_rows_f32(fc_w +  0 * 64 + 32, 64); f0 = wmma_f16(a1, bw, f0);
        bw = load_B_rows_f32(fc_w + 16 * 64,      64); f1 = wmma_f16(a0, bw, f1);
        bw = load_B_rows_f32(fc_w + 16 * 64 + 32, 64); f1 = wmma_f16(a1, bw, f1);
        bw = load_B_rows_f32(fc_w + 32 * 64,      64); f2 = wmma_f16(a0, bw, f2);
        bw = load_B_rows_f32(fc_w + 32 * 64 + 32, 64); f2 = wmma_f16(a1, bw, f2);
        bw = load_B_rows_f32(fc_w + 48 * 64,      64); f3 = wmma_f16(a0, bw, f3);
        bw = load_B_rows_f32(fc_w + 48 * 64 + 32, 64); f3 = wmma_f16(a1, bw, f3);
    }

    // ---- write output ----
    size_t rowbase = (size_t)(b * NV_ + q0);
#pragma unroll
    for (int j = 0; j < 8; ++j) {
        size_t ro = (rowbase + rbase + j) * 128;
        out[ro +  64 + col] = fmaxf(f0[j], 0.0f);
        out[ro +  80 + col] = fmaxf(f1[j], 0.0f);
        out[ro +  96 + col] = fmaxf(f2[j], 0.0f);
        out[ro + 112 + col] = fmaxf(f3[j], 0.0f);
    }
    const float* vfb = voxel_features + rowbase * C_;
#pragma unroll
    for (int i = 0; i < 8; ++i) {
        int f   = lane + 32 * i;
        int row = f >> 4;
        int c4  = f & 15;
        float4 val = ((const float4*)vfb)[row * 16 + c4];
        *(float4*)(out + (rowbase + row) * 128 + c4 * 4) = val;
    }
}

// ---------------------------------------------------------------------------
extern "C" void kernel_launch(void* const* d_in, const int* in_sizes, int n_in,
                              void* d_out, int out_size, void* d_ws, size_t ws_size,
                              hipStream_t stream) {
    const float* voxel_features = (const float*)d_in[0];
    const float* voxels_3d      = (const float*)d_in[1];
    const float* img_feats      = (const float*)d_in[2];
    const int*   vox_x          = (const int*)d_in[3];
    const int*   vox_y          = (const int*)d_in[4];
    const float* q_w            = (const float*)d_in[5];
    const float* q_b            = (const float*)d_in[6];
    const float* k_w            = (const float*)d_in[7];
    const float* k_b            = (const float*)d_in[8];
    const float* v_w            = (const float*)d_in[9];
    const float* v_b            = (const float*)d_in[10];
    const float* o_w            = (const float*)d_in[11];
    const float* o_b            = (const float*)d_in[12];
    const float* fc_w           = (const float*)d_in[13];
    float* out = (float*)d_out;

    const size_t elems = (size_t)B_ * NV_ * C_;
    half_t* Qh = (half_t*)d_ws;
    half_t* Kh = Qh + elems;
    half_t* Vt = Kh + elems;                      // 6 MB of workspace total

    qkv_prologue<<<(B_ * NV_) / 4, 256, 0, stream>>>(
        voxel_features, voxels_3d, img_feats, vox_x, vox_y,
        q_w, q_b, k_w, k_b, v_w, v_b, Qh, Kh, Vt);

    flash_attn_epilogue<<<(B_ * NV_) / 64, 128, 0, stream>>>(
        Qh, Kh, Vt, voxel_features, o_w, o_b, fc_w, out);
}
